// GCN_31164282700070
// MI455X (gfx1250) — matrix operand order repaired and checked by reading, not compiled, not run on hardware
//
#include <hip/hip_runtime.h>
#include <hip/hip_bf16.h>

typedef __attribute__((ext_vector_type(16))) __bf16 v16bf;
typedef __attribute__((ext_vector_type(16))) float  v16f;
typedef __attribute__((ext_vector_type(8)))  float  v8f;

#define GCN_N_NODES 100000
#define GCN_N_EDGES 1250000
#define GCN_IN_CH   128
#define GCN_HID_CH  64
#define GCN_CLS_CH  32

__device__ __forceinline__ float f32_atomic_add(float* p, float v) {
    return __hip_atomic_fetch_add(p, v, __ATOMIC_RELAXED, __HIP_MEMORY_SCOPE_AGENT);
}

// ---------------------------------------------------------------------------
// Prep: N-major bf16 copies of the weights, so each WMMA B operand is one
// contiguous 32B load. W1t[n][k] (64x128 bf16 = 16KB), W2t[n][k] (32x64 = 4KB).
// ---------------------------------------------------------------------------
__global__ __launch_bounds__(256) void gcn_prep_w(const float* __restrict__ W1,
                                                  const float* __restrict__ W2,
                                                  __bf16* __restrict__ W1t,
                                                  __bf16* __restrict__ W2t) {
    const int t = blockIdx.x * 256 + threadIdx.x;
    if (t < GCN_HID_CH * GCN_IN_CH) {            // 8192
        const int n = t >> 7, k = t & (GCN_IN_CH - 1);
        W1t[t] = (__bf16)W1[k * GCN_HID_CH + n];
    }
    if (t < GCN_CLS_CH * GCN_HID_CH) {           // 2048
        const int n = t >> 6, k = t & (GCN_HID_CH - 1);
        W2t[t] = (__bf16)W2[k * GCN_CLS_CH + n];
    }
}

// ---------------------------------------------------------------------------
// GEMM1: H1[100000,64] = X[100000,128] @ W1[128,64]   (bf16 WMMA, f32 accum)
// 8 waves per block, 16 rows per wave -> 128 rows per block.
// ---------------------------------------------------------------------------
__global__ __launch_bounds__(256) void gcn_gemm1(const float* __restrict__ X,
                                                 const __bf16* __restrict__ W1t,
                                                 float* __restrict__ H1) {
    const int lane = threadIdx.x & 31;
    const int wave = threadIdx.x >> 5;
    const int rowBase = blockIdx.x * 128 + wave * 16;
    const int ml = lane & 15;
    const int hi = lane >> 4;

    int arow = rowBase + ml;
    if (arow >= GCN_N_NODES) arow = GCN_N_NODES - 1;   // clamp, keep EXEC full
    const float* xr = X + arow * GCN_IN_CH;

    v8f acc[4];
    #pragma unroll
    for (int nb = 0; nb < 4; ++nb) acc[nb] = (v8f){0.f,0.f,0.f,0.f,0.f,0.f,0.f,0.f};

    #pragma unroll
    for (int kb = 0; kb < GCN_IN_CH; kb += 32) {
        // A operand: elem i <- K = kb+16*hi+i
        const float4 p0 = *(const float4*)(xr + kb + 16 * hi);
        const float4 p1 = *(const float4*)(xr + kb + 16 * hi + 4);
        const float4 p2 = *(const float4*)(xr + kb + 16 * hi + 8);
        const float4 p3 = *(const float4*)(xr + kb + 16 * hi + 12);
        const v16f af = {p0.x, p0.y, p0.z, p0.w, p1.x, p1.y, p1.z, p1.w,
                         p2.x, p2.y, p2.z, p2.w, p3.x, p3.y, p3.z, p3.w};
        const v16bf a = __builtin_convertvector(af, v16bf);

        #pragma unroll
        for (int nb = 0; nb < 4; ++nb) {
            const int n = nb * 16 + ml;
            // B operand: elem i <- K = kb + 16*hi + i, col n  (contiguous in W1t)
            const v16bf b = *(const v16bf*)(W1t + n * GCN_IN_CH + kb + 16 * hi);
            acc[nb] = __builtin_amdgcn_wmma_f32_16x16x32_bf16(
                false, a, false, b, (short)0, acc[nb], false, false);
        }
    }

    // D layout: elem r at lane -> (m = 8*hi + r, n = ml)
    const int mrow = rowBase + 8 * hi;
    #pragma unroll
    for (int nb = 0; nb < 4; ++nb) {
        #pragma unroll
        for (int r = 0; r < 8; ++r) {
            const int row = mrow + r;
            if (row < GCN_N_NODES)
                H1[row * GCN_HID_CH + nb * 16 + ml] = acc[nb][r];
        }
    }
}

// ---------------------------------------------------------------------------
// Edge aggregation, 64 channels: A1[dst] += H1[src]; 16 threads per edge,
// float4 gather (coalesced 256B per edge) + 4 f32 atomics into L2-resident A1.
// ---------------------------------------------------------------------------
__global__ __launch_bounds__(256) void gcn_agg64(const float* __restrict__ H,
                                                 const int* __restrict__ src,
                                                 const int* __restrict__ dst,
                                                 float* __restrict__ A) {
    const int t = blockIdx.x * 256 + threadIdx.x;
    if (t >= GCN_N_EDGES * 16) return;
    const int e = t >> 4, q = t & 15;
    const int s = src[e], d = dst[e];
    const float4 v = *(const float4*)(H + s * GCN_HID_CH + q * 4);
    float* o = A + d * GCN_HID_CH + q * 4;
    f32_atomic_add(o + 0, v.x);
    f32_atomic_add(o + 1, v.y);
    f32_atomic_add(o + 2, v.z);
    f32_atomic_add(o + 3, v.w);
}

// ---------------------------------------------------------------------------
// GEMM2: H2[100000,32] = relu(A1 + b1) @ W2[64,32]   (bias+ReLU fused in load)
// ---------------------------------------------------------------------------
__global__ __launch_bounds__(256) void gcn_gemm2(const float* __restrict__ A1,
                                                 const float* __restrict__ b1,
                                                 const __bf16* __restrict__ W2t,
                                                 float* __restrict__ H2) {
    const int lane = threadIdx.x & 31;
    const int wave = threadIdx.x >> 5;
    const int rowBase = blockIdx.x * 128 + wave * 16;
    const int ml = lane & 15;
    const int hi = lane >> 4;

    int arow = rowBase + ml;
    if (arow >= GCN_N_NODES) arow = GCN_N_NODES - 1;
    const float* ar = A1 + arow * GCN_HID_CH;

    v8f acc[2];
    acc[0] = (v8f){0.f,0.f,0.f,0.f,0.f,0.f,0.f,0.f};
    acc[1] = acc[0];

    #pragma unroll
    for (int kb = 0; kb < GCN_HID_CH; kb += 32) {
        const int k0 = kb + 16 * hi;      // elems 0..7
        const int k1 = kb + 16 * hi + 8;  // elems 8..15
        const float4 p0 = *(const float4*)(ar + k0);
        const float4 p1 = *(const float4*)(ar + k0 + 4);
        const float4 p2 = *(const float4*)(ar + k1);
        const float4 p3 = *(const float4*)(ar + k1 + 4);
        const float4 q0 = *(const float4*)(b1 + k0);
        const float4 q1 = *(const float4*)(b1 + k0 + 4);
        const float4 q2 = *(const float4*)(b1 + k1);
        const float4 q3 = *(const float4*)(b1 + k1 + 4);
        v16f af = {p0.x + q0.x, p0.y + q0.y, p0.z + q0.z, p0.w + q0.w,
                   p1.x + q1.x, p1.y + q1.y, p1.z + q1.z, p1.w + q1.w,
                   p2.x + q2.x, p2.y + q2.y, p2.z + q2.z, p2.w + q2.w,
                   p3.x + q3.x, p3.y + q3.y, p3.z + q3.z, p3.w + q3.w};
        #pragma unroll
        for (int i = 0; i < 16; ++i) af[i] = fmaxf(af[i], 0.f);   // fused ReLU
        const v16bf a = __builtin_convertvector(af, v16bf);

        #pragma unroll
        for (int nb = 0; nb < 2; ++nb) {
            const int n = nb * 16 + ml;
            const v16bf b = *(const v16bf*)(W2t + n * GCN_HID_CH + kb + 16 * hi);
            acc[nb] = __builtin_amdgcn_wmma_f32_16x16x32_bf16(
                false, a, false, b, (short)0, acc[nb], false, false);
        }
    }

    const int mrow = rowBase + 8 * hi;
    #pragma unroll
    for (int nb = 0; nb < 2; ++nb) {
        #pragma unroll
        for (int r = 0; r < 8; ++r) {
            const int row = mrow + r;
            if (row < GCN_N_NODES)
                H2[row * GCN_CLS_CH + nb * 16 + ml] = acc[nb][r];
        }
    }
}

// ---------------------------------------------------------------------------
// Edge aggregation, 32 channels: out[dst] += H2[src]; 8 threads per edge.
// ---------------------------------------------------------------------------
__global__ __launch_bounds__(256) void gcn_agg32(const float* __restrict__ H,
                                                 const int* __restrict__ src,
                                                 const int* __restrict__ dst,
                                                 float* __restrict__ A) {
    const int t = blockIdx.x * 256 + threadIdx.x;
    if (t >= GCN_N_EDGES * 8) return;
    const int e = t >> 3, q = t & 7;
    const int s = src[e], d = dst[e];
    const float4 v = *(const float4*)(H + s * GCN_CLS_CH + q * 4);
    float* o = A + d * GCN_CLS_CH + q * 4;
    f32_atomic_add(o + 0, v.x);
    f32_atomic_add(o + 1, v.y);
    f32_atomic_add(o + 2, v.z);
    f32_atomic_add(o + 3, v.w);
}

// out = sigmoid(out + b2), in place
__global__ __launch_bounds__(256) void gcn_sigmoid(float* __restrict__ out,
                                                   const float* __restrict__ b2) {
    const int t = blockIdx.x * 256 + threadIdx.x;
    if (t >= GCN_N_NODES * GCN_CLS_CH) return;
    const float v = out[t] + b2[t & (GCN_CLS_CH - 1)];
    out[t] = 1.0f / (1.0f + __expf(-v));
}

extern "C" void kernel_launch(void* const* d_in, const int* in_sizes, int n_in,
                              void* d_out, int out_size, void* d_ws, size_t ws_size,
                              hipStream_t stream) {
    const float* X  = (const float*)d_in[0];
    const int*   ei = (const int*)  d_in[1];   // [2, E]: row0 = src, row1 = dst
    const float* W1 = (const float*)d_in[2];
    const float* b1 = (const float*)d_in[3];
    const float* W2 = (const float*)d_in[4];
    const float* b2 = (const float*)d_in[5];
    float* out = (float*)d_out;

    char* ws = (char*)d_ws;
    float*  H1  = (float*)(ws);                       // 25,600,000 B
    float*  A1  = (float*)(ws + 25600000);            // 25,600,000 B
    float*  H2  = (float*)(ws + 51200000);            // 12,800,000 B
    __bf16* W1t = (__bf16*)(ws + 64000000);           // 16,384 B
    __bf16* W2t = (__bf16*)(ws + 64016384);           //  4,096 B

    // zero the atomic accumulators every call (graph-capture safe, deterministic)
    hipMemsetAsync(A1,  0, (size_t)GCN_N_NODES * GCN_HID_CH * sizeof(float), stream);
    hipMemsetAsync(out, 0, (size_t)GCN_N_NODES * GCN_CLS_CH * sizeof(float), stream);

    gcn_prep_w<<<(GCN_HID_CH * GCN_IN_CH + 255) / 256, 256, 0, stream>>>(W1, W2, W1t, W2t);

    const int nBlocksRows = (GCN_N_NODES + 127) / 128;
    gcn_gemm1<<<nBlocksRows, 256, 0, stream>>>(X, W1t, H1);
    gcn_agg64<<<(GCN_N_EDGES * 16 + 255) / 256, 256, 0, stream>>>(H1, ei, ei + GCN_N_EDGES, A1);
    gcn_gemm2<<<nBlocksRows, 256, 0, stream>>>(A1, b1, W2t, H2);
    gcn_agg32<<<(GCN_N_EDGES * 8 + 255) / 256, 256, 0, stream>>>(H2, ei, ei + GCN_N_EDGES, out);
    gcn_sigmoid<<<(GCN_N_NODES * GCN_CLS_CH + 255) / 256, 256, 0, stream>>>(out, b2);
}
